// MultiHeadAttention_22144851378311
// MI455X (gfx1250) — matrix operand order repaired and checked by reading, not compiled, not run on hardware
//
#include <hip/hip_runtime.h>
#include <cstdint>
#include <cstddef>

// ---------------------------------------------------------------------------
// Problem dims (fixed by the reference)
// ---------------------------------------------------------------------------
constexpr int Bb  = 8;
constexpr int Ll  = 512;
constexpr int Dd  = 1024;
constexpr int Hh  = 16;
constexpr int PKk = 5;
constexpr int DKk = 64;
constexpr int Mrows = Bb * Ll;   // 4096 rows for all D x D GEMMs

#define DEV __device__ __forceinline__

typedef __attribute__((ext_vector_type(16))) __bf16   v16bf;
typedef __attribute__((ext_vector_type(8)))  __bf16   v8bf;
typedef __attribute__((ext_vector_type(8)))  float    v8f;
typedef __attribute__((ext_vector_type(4)))  uint32_t u32x4;
typedef __attribute__((ext_vector_type(8)))  uint32_t u32x8;

// ---------------------------------------------------------------------------
// WMMA fragment helpers (CDNA5 wave32 layouts, CDNA5 ISA 7.12.2)
//
// A (16x32 bf16): lane (l&15) = row M; half = l>>4 selects K sub-block.
//   elements 0..7  = K = half*8 + 0..7
//   elements 8..15 = K = half*8 + 16..23
// B (32x16 bf16): lane (l&15) = column N; half = l>>4.
//   elements 0..7  = K = half*8 + 0..7;  elements 8..15 = K = half*8 + 16..23
// C/D (16x16 f32): VGPR r, lanes 0-15: (M=r, N=lane); lanes 16-31: (M=8+r).
// ---------------------------------------------------------------------------
DEV v16bf ld_a32(const __bf16* p) {  // p -> this lane's first 8 elems; next 8 at p+16
  v8bf lo = *(const v8bf*)p;
  v8bf hi = *(const v8bf*)(p + 16);
  v16bf r;
#pragma unroll
  for (int i = 0; i < 8; ++i) { r[i] = lo[i]; r[i + 8] = hi[i]; }
  return r;
}

DEV v16bf ld_b32(const __bf16* p) {  // p -> this lane's first 8 elems; next 8 at p+16
  v8bf lo = *(const v8bf*)p;
  v8bf hi = *(const v8bf*)(p + 16);
  v16bf r;
#pragma unroll
  for (int i = 0; i < 8; ++i) { r[i] = lo[i]; r[i + 8] = hi[i]; }
  return r;
}

DEV v8f wmma_bf16(v16bf a, v16bf b, v8f c) {
  return __builtin_amdgcn_wmma_f32_16x16x32_bf16(
      /*neg_a=*/false, a, /*neg_b=*/false, b,
      /*c_mod=*/(short)0, c, /*reuse_a=*/false, /*reuse_b=*/false);
}

// LDS byte offset of a __shared__ object (ptrtoint of addrspace(3) pointer)
template <typename T>
DEV unsigned lds_offset(T* p) {
  return (unsigned)(uintptr_t)(__attribute__((address_space(3))) T*)p;
}

// ---------------------------------------------------------------------------
// TDM: DMA a 64-row x 32-col bf16 tile of a row-major [*, 1024] tensor into
// LDS (contiguous [64][32]).  2-D tensor -> 2 descriptor groups (D# per
// CDNA5 ISA).  Tracked by TENSORcnt.
// ---------------------------------------------------------------------------
DEV void tdm_load_w_tile(unsigned lds_byte_off, const __bf16* gsrc) {
  const uint64_t ga = (uint64_t)(uintptr_t)gsrc;
  u32x4 g0;
  g0[0] = 1u;                                   // count=1, is_restore=0, gather=0
  g0[1] = lds_byte_off;                         // lds_addr (bytes)
  g0[2] = (uint32_t)ga;                         // global_addr[31:0]
  g0[3] = ((uint32_t)(ga >> 32) & 0x01FFFFFFu)  // global_addr[56:32]
          | (2u << 30);                         // type = 2 ("image")
  u32x8 g1;
  g1[0] = 0x00010000u;                          // wg_mask=0, data_size=1 (2 bytes)
  g1[1] = ((uint32_t)Dd & 0xFFFFu) << 16;       // tensor_dim0[15:0] = 1024
  g1[2] = ((uint32_t)Dd >> 16)                  // tensor_dim0[31:16]
          | (64u << 16);                        // tensor_dim1[15:0] = 64 rows
  g1[3] = (32u << 16);                          // tensor_dim1[31:16]=0 | tile_dim0=32
  g1[4] = 64u;                                  // tile_dim1=64, tile_dim2=0
  g1[5] = (uint32_t)Dd;                         // tensor_dim0_stride[31:0] = 1024
  g1[6] = 0u;                                   // stride hi, dim1_stride lo (unused 2D)
  g1[7] = 0u;
  asm volatile("tensor_load_to_lds %0, %1" :: "s"(g0), "s"(g1) : "memory");
}

// ---------------------------------------------------------------------------
// f32 -> bf16 weight convert
// ---------------------------------------------------------------------------
__global__ void cvt_bf16_kernel(const float* __restrict__ x, __bf16* __restrict__ y, int n) {
  int i = blockIdx.x * blockDim.x + threadIdx.x;
  if (i < n) y[i] = (__bf16)x[i];
}

// ---------------------------------------------------------------------------
// LayerNorm over last dim (D=1024), output bf16. One block per row.
// ---------------------------------------------------------------------------
__global__ __launch_bounds__(256)
void layernorm_kernel(const float* __restrict__ x, const float* __restrict__ g,
                      const float* __restrict__ b, __bf16* __restrict__ y) {
  __shared__ float ssum[256];
  __shared__ float ssq[256];
  const int row = blockIdx.x;
  const float* xr = x + (size_t)row * Dd;
  float s = 0.f, q = 0.f;
  for (int i = threadIdx.x; i < Dd; i += 256) { float v = xr[i]; s += v; q += v * v; }
  ssum[threadIdx.x] = s; ssq[threadIdx.x] = q;
  __syncthreads();
  for (int st = 128; st > 0; st >>= 1) {
    if (threadIdx.x < st) { ssum[threadIdx.x] += ssum[threadIdx.x + st];
                            ssq[threadIdx.x]  += ssq[threadIdx.x + st]; }
    __syncthreads();
  }
  const float mu  = ssum[0] * (1.0f / Dd);
  const float var = ssq[0] * (1.0f / Dd) - mu * mu;
  const float rs  = rsqrtf(var + 1e-6f);
  for (int i = threadIdx.x; i < Dd; i += 256)
    y[(size_t)row * Dd + i] = (__bf16)((xr[i] - mu) * rs * g[i] + b[i]);
}

// ---------------------------------------------------------------------------
// GEMM: Y[M,N] = act(A[M,K] @ W[N,K]^T + bias)   (M=4096, N=K=1024, bf16 in)
// 128 threads = 4 waves; block tile 64x64.  W tile (64n x 32k) is DMA'd into
// double-buffered LDS by the Tensor Data Mover (wave 0 issues, TENSORcnt
// waits, workgroup barrier publishes), overlapping DMA(k+32) with WMMA(k).
// ACT: 0 = none, 1 = relu, 2 = sigmoid
// ---------------------------------------------------------------------------
template <int ACT>
__global__ __launch_bounds__(128)
void gemm_bf16_kernel(const __bf16* __restrict__ A, const __bf16* __restrict__ W,
                      const float* __restrict__ bias,
                      float* __restrict__ Yf, __bf16* __restrict__ Ybf) {
  constexpr int K = Dd, N = Dd;
  __shared__ __bf16 Wt[2][64 * 32];  // [buf][n][k] row-major, 2 x 4 KB

  const int tid  = threadIdx.x;
  const int lane = tid & 31;
  const int wave = tid >> 5;
  const int hf   = lane >> 4;   // half-wave selector
  const int ln   = lane & 15;
  const int row0 = blockIdx.y * 64 + wave * 16;
  const int col0 = blockIdx.x * 64;

  v8f acc[4] = {v8f{}, v8f{}, v8f{}, v8f{}};

  const unsigned ldsOff0 = lds_offset(&Wt[0][0]);
  const unsigned ldsOff1 = lds_offset(&Wt[1][0]);
  const __bf16* wbase = W + (size_t)col0 * K;

  if (wave == 0) tdm_load_w_tile(ldsOff0, wbase);  // preload k0 = 0 into buf 0

  int cur = 0;
  for (int k0 = 0; k0 < K; k0 += 32) {
    if (wave == 0) {
      const bool more = (k0 + 32) < K;
      if (more) tdm_load_w_tile(cur ? ldsOff0 : ldsOff1, wbase + (k0 + 32));
      if (more) __builtin_amdgcn_s_wait_tensorcnt(1);  // current tile complete
      else      __builtin_amdgcn_s_wait_tensorcnt(0);
    }
    __syncthreads();  // publish DMA'd tile to all waves

    // A fragment: row (row0+ln), K chunk starts at k0 + hf*8 (+16 for 2nd half)
    v16bf afrag = ld_a32(A + (size_t)(row0 + ln) * K + k0 + hf * 8);

#pragma unroll
    for (int nt = 0; nt < 4; ++nt) {
      v16bf bfrag = ld_b32(&Wt[cur][(nt * 16 + ln) * 32 + hf * 8]);
      acc[nt] = wmma_bf16(afrag, bfrag, acc[nt]);
    }
    __syncthreads();  // all reads of Wt[cur] done before it is re-DMA'd
    cur ^= 1;
  }

#pragma unroll
  for (int nt = 0; nt < 4; ++nt) {
    const int col = col0 + nt * 16 + ln;
    const float bv = bias ? bias[col] : 0.0f;
#pragma unroll
    for (int r = 0; r < 8; ++r) {
      const int row = row0 + r + hf * 8;
      float v = acc[nt][r] + bv;
      if (ACT == 1) v = fmaxf(v, 0.0f);
      if (ACT == 2) v = 1.0f / (1.0f + __expf(-v));
      if (Yf)  Yf[(size_t)row * N + col] = v;
      if (Ybf) Ybf[(size_t)row * N + col] = (__bf16)v;
    }
  }
}

// ---------------------------------------------------------------------------
// V transpose: [B,L,D] bf16 -> [B,H,DK,L] bf16 (so PV B-fragments are contiguous)
// ---------------------------------------------------------------------------
__global__ void transpose_v_kernel(const __bf16* __restrict__ v, __bf16* __restrict__ vT) {
  const int idx = blockIdx.x * blockDim.x + threadIdx.x;  // over B*L*D
  const int dd  = idx % Dd;
  const int rem = idx / Dd;
  const int j   = rem % Ll;
  const int b   = rem / Ll;
  const int h   = dd >> 6;
  const int d   = dd & 63;
  vT[(((size_t)b * Hh + h) * DKk + d) * Ll + j] = v[idx];
}

// ---------------------------------------------------------------------------
// qr[b,h,i,p] = sum_d q[b,i,h*64+d] * rel_k[p,d]   (one thread per (b,h,i))
// ---------------------------------------------------------------------------
__global__ __launch_bounds__(256)
void qr_kernel(const __bf16* __restrict__ qb, const float* __restrict__ rel_k,
               float* __restrict__ qr) {
  const int t = blockIdx.x * blockDim.x + threadIdx.x;  // over B*H*L
  const int i   = t % Ll;
  const int rem = t / Ll;
  const int h   = rem % Hh;
  const int b   = rem / Hh;
  const __bf16* qp = qb + ((size_t)b * Ll + i) * Dd + h * 64;
#pragma unroll
  for (int p = 0; p < PKk; ++p) {
    float acc = 0.f;
#pragma unroll 8
    for (int d = 0; d < DKk; ++d) acc += (float)qp[d] * rel_k[p * DKk + d];
    qr[(size_t)t * PKk + p] = acc;
  }
}

// ---------------------------------------------------------------------------
// Attention core: one wave per (b, h, 16-row query tile).
//   S = Q K^T  (2 WMMAs / 16-key tile) + rel bias, scale, mask -> LDS f32
//   row softmax (shfl_xor(16) cross-half reduce), P stored bf16 in LDS
//   O = P V    (16 k-steps x 4 n-tiles of WMMA), scaled by 1/rowsum
// ---------------------------------------------------------------------------
__global__ __launch_bounds__(32)
void attn_kernel(const __bf16* __restrict__ qb, const __bf16* __restrict__ kb,
                 const __bf16* __restrict__ vT, const float* __restrict__ qr,
                 const int* __restrict__ gpm, const unsigned char* __restrict__ mask,
                 float* __restrict__ ao) {
  __shared__ float  Ssc[16 * Ll];      // 32 KB scores
  __shared__ __bf16 Pp[16 * Ll];       // 16 KB probabilities (bf16)
  __shared__ float  qrl[16 * PKk];
  __shared__ float  inv_sum[16];

  const int lane = threadIdx.x;
  const int hf   = lane >> 4;
  const int ln   = lane & 15;
  const int i0   = blockIdx.x * 16;
  const int h    = blockIdx.y;
  const int b    = blockIdx.z;

  // stage the 16x5 qr tile
  for (int idx = lane; idx < 16 * PKk; idx += 32) {
    const int m = idx / PKk, p = idx % PKk;
    qrl[idx] = qr[(((size_t)b * Hh + h) * Ll + i0 + m) * PKk + p];
  }
  __syncthreads();

  // Q fragments (K = 0..31 and 32..63 of the head dim), loaded once
  const __bf16* qrow = qb + ((size_t)b * Ll + i0 + ln) * Dd + h * 64;
  const v16bf aq0 = ld_a32(qrow + hf * 8);
  const v16bf aq1 = ld_a32(qrow + 32 + hf * 8);

  // ---- pass 1: scores ----
  for (int j0 = 0; j0 < Ll; j0 += 16) {
    const __bf16* kp = kb + ((size_t)b * Ll + j0 + ln) * Dd + h * 64 + hf * 8;
    const v16bf bk0 = ld_b32(kp);
    const v16bf bk1 = ld_b32(kp + 32);
    v8f s = v8f{};
    s = wmma_bf16(aq0, bk0, s);
    s = wmma_bf16(aq1, bk1, s);

    const int j = j0 + ln;
    const unsigned char mk = mask[b * Ll + j];
#pragma unroll
    for (int r = 0; r < 8; ++r) {
      const int m = r + hf * 8;
      const int i = i0 + m;
      const int p = gpm[((size_t)b * Ll + i) * Ll + j];
      float val = (s[r] + qrl[m * PKk + p]) * 0.125f;  // 1/sqrt(64)
      if (!mk) val = -9.0e9f;
      Ssc[m * Ll + j] = val;
    }
  }
  __syncthreads();

  // ---- pass 2: softmax (lane l handles half hf of row ln) ----
  {
    const int m = ln;
    const int base = hf * (Ll / 2);
    float mx = -3.4e38f;
    for (int j = 0; j < Ll / 2; ++j) mx = fmaxf(mx, Ssc[m * Ll + base + j]);
    mx = fmaxf(mx, __shfl_xor(mx, 16, 32));
    float sm = 0.f;
    for (int j = 0; j < Ll / 2; ++j) {
      const float e = __expf(Ssc[m * Ll + base + j] - mx);
      Pp[m * Ll + base + j] = (__bf16)e;
      sm += e;
    }
    sm += __shfl_xor(sm, 16, 32);
    if (hf == 0) inv_sum[m] = 1.0f / sm;
  }
  __syncthreads();

  // ---- pass 3: O = P @ V ----
  v8f o[4] = {v8f{}, v8f{}, v8f{}, v8f{}};
  for (int kk = 0; kk < Ll; kk += 32) {
    const v16bf ap = ld_a32(&Pp[ln * Ll + kk + hf * 8]);
#pragma unroll
    for (int nt = 0; nt < 4; ++nt) {
      const __bf16* vp = vT + (((size_t)b * Hh + h) * DKk + nt * 16 + ln) * Ll + kk + hf * 8;
      o[nt] = wmma_bf16(ap, ld_b32(vp), o[nt]);
    }
  }

#pragma unroll
  for (int nt = 0; nt < 4; ++nt) {
#pragma unroll
    for (int r = 0; r < 8; ++r) {
      const int m = r + hf * 8;
      ao[((size_t)b * Ll + i0 + m) * Dd + h * 64 + nt * 16 + ln] = o[nt][r] * inv_sum[m];
    }
  }
}

// ---------------------------------------------------------------------------
// z = bf16(ao * gate)
// ---------------------------------------------------------------------------
__global__ void fuse_gate_kernel(const float* __restrict__ ao, const float* __restrict__ gate,
                                 __bf16* __restrict__ z) {
  const int i = blockIdx.x * blockDim.x + threadIdx.x;
  z[i] = (__bf16)(ao[i] * gate[i]);
}

// ---------------------------------------------------------------------------
// host: workspace carve-up + launches (all on `stream`, graph-capture safe)
// ---------------------------------------------------------------------------
extern "C" void kernel_launch(void* const* d_in, const int* in_sizes, int n_in,
                              void* d_out, int out_size, void* d_ws, size_t ws_size,
                              hipStream_t stream) {
  (void)in_sizes; (void)n_in; (void)out_size; (void)ws_size;

  const float* src    = (const float*)d_in[0];
  const float* tgt    = (const float*)d_in[1];
  const int*   gpm    = (const int*)d_in[2];
  const unsigned char* mask = (const unsigned char*)d_in[3];
  const float* ln_g   = (const float*)d_in[4];
  const float* ln_b   = (const float*)d_in[5];
  const float* q_w    = (const float*)d_in[6];
  const float* k_w    = (const float*)d_in[7];
  const float* v_w1   = (const float*)d_in[8];
  const float* v_b1   = (const float*)d_in[9];
  const float* v_w2   = (const float*)d_in[10];
  const float* v_b2   = (const float*)d_in[11];
  const float* rel_k  = (const float*)d_in[12];
  const float* gate_w = (const float*)d_in[13];
  const float* gate_b = (const float*)d_in[14];
  const float* out_w  = (const float*)d_in[15];
  const float* out_b  = (const float*)d_in[16];
  float* out = (float*)d_out;

  char* ws = (char*)d_ws;
  size_t off = 0;
  auto alloc = [&](size_t bytes) -> char* {
    char* p = ws + off;
    off = (off + bytes + 255) & ~(size_t)255;
    return p;
  };

  const size_t actBf = (size_t)Mrows * Dd * sizeof(__bf16);   // 8 MB
  const size_t actF  = (size_t)Mrows * Dd * sizeof(float);    // 16 MB
  const size_t wBf   = (size_t)Dd * Dd * sizeof(__bf16);      // 2 MB

  __bf16* qw_bf  = (__bf16*)alloc(wBf);
  __bf16* kw_bf  = (__bf16*)alloc(wBf);
  __bf16* v1w_bf = (__bf16*)alloc(wBf);
  __bf16* v2w_bf = (__bf16*)alloc(wBf);
  __bf16* gw_bf  = (__bf16*)alloc(wBf);
  __bf16* ow_bf  = (__bf16*)alloc(wBf);
  __bf16* s_bf   = (__bf16*)alloc(actBf);
  __bf16* t_bf   = (__bf16*)alloc(actBf);
  __bf16* q_bf   = (__bf16*)alloc(actBf);
  __bf16* k_bf   = (__bf16*)alloc(actBf);
  __bf16* h1_bf  = (__bf16*)alloc(actBf);
  __bf16* v_bf   = (__bf16*)alloc(actBf);
  __bf16* vT_bf  = (__bf16*)alloc(actBf);
  float*  qr_f   = (float*)alloc((size_t)Bb * Hh * Ll * PKk * sizeof(float));
  float*  gate_f = (float*)alloc(actF);
  float*  ao_f   = (float*)alloc(actF);
  __bf16* z_bf   = (__bf16*)alloc(actBf);

  const int nW = Dd * Dd;
  const dim3 cvtGrid((nW + 255) / 256);
  cvt_bf16_kernel<<<cvtGrid, 256, 0, stream>>>(q_w,    qw_bf,  nW);
  cvt_bf16_kernel<<<cvtGrid, 256, 0, stream>>>(k_w,    kw_bf,  nW);
  cvt_bf16_kernel<<<cvtGrid, 256, 0, stream>>>(v_w1,   v1w_bf, nW);
  cvt_bf16_kernel<<<cvtGrid, 256, 0, stream>>>(v_w2,   v2w_bf, nW);
  cvt_bf16_kernel<<<cvtGrid, 256, 0, stream>>>(gate_w, gw_bf,  nW);
  cvt_bf16_kernel<<<cvtGrid, 256, 0, stream>>>(out_w,  ow_bf,  nW);

  layernorm_kernel<<<Mrows, 256, 0, stream>>>(src, ln_g, ln_b, s_bf);
  layernorm_kernel<<<Mrows, 256, 0, stream>>>(tgt, ln_g, ln_b, t_bf);

  const dim3 gGrid(Dd / 64, Mrows / 64);   // (16, 64)
  gemm_bf16_kernel<0><<<gGrid, 128, 0, stream>>>(s_bf,  kw_bf,  nullptr, nullptr, k_bf);
  gemm_bf16_kernel<0><<<gGrid, 128, 0, stream>>>(t_bf,  qw_bf,  nullptr, nullptr, q_bf);
  gemm_bf16_kernel<1><<<gGrid, 128, 0, stream>>>(t_bf,  v1w_bf, v_b1,    nullptr, h1_bf);
  gemm_bf16_kernel<0><<<gGrid, 128, 0, stream>>>(h1_bf, v2w_bf, v_b2,    nullptr, v_bf);
  gemm_bf16_kernel<2><<<gGrid, 128, 0, stream>>>(s_bf,  gw_bf,  gate_b,  gate_f,  nullptr);

  transpose_v_kernel<<<(Mrows * Dd) / 256, 256, 0, stream>>>(v_bf, vT_bf);
  qr_kernel<<<(Bb * Hh * Ll) / 256, 256, 0, stream>>>(q_bf, rel_k, qr_f);

  const dim3 aGrid(Ll / 16, Hh, Bb);       // (32, 16, 8)
  attn_kernel<<<aGrid, 32, 0, stream>>>(q_bf, k_bf, vT_bf, qr_f, gpm, mask, ao_f);

  fuse_gate_kernel<<<(Mrows * Dd) / 256, 256, 0, stream>>>(ao_f, gate_f, z_bf);

  gemm_bf16_kernel<0><<<gGrid, 128, 0, stream>>>(z_bf, ow_bf, out_b, out, nullptr);
}
